// HMM_35837207118423
// MI455X (gfx1250) — matrix operand order repaired and checked by reading, not compiled, not run on hardware
//
#include <hip/hip_runtime.h>
#include <math.h>

// ---------------------------------------------------------------------------
// HMM forward (512 states, 510 steps) recast as a binary tree of bf16 WMMA
// GEMMs on gfx1250.
//
//   reference scan:  alpha <- lt + logsumexp(alpha + logT[t], axis=1)
//   linear space  :  a <- diag(e^{lt}) * softmax_rows(W_t) * a
//   => a_final = (M_510 * ... * M_1) * a0   (associative -> tree reduction)
//
// Layout rule: in P_p = M[2p+1] x M[2p], the left operand is consumed
// row-major (A fragments) and the right operand column-major (B fragments).
// Slot parity is known at production time, so every matrix is written ONCE in
// exactly the layout its consumer wants:
//    odd  slot -> row-major    (storage[i*512+j] = M[i][j])
//    even slot -> column-major (storage[j*512+i] = M[i][j])
// This removes all scalar LDS transpose stores from the GEMM hot loop: both
// operands stage with global_load_b128 -> ds_store_b128 -> ds_load_b128.
//
// Each matrix is normalized (entries <= 1) with a scalar log-scale on the
// side; products renormalize by the provable bound 512*bound_L*bound_R and
// recover exact bounds with a deterministic monotone-uint atomicMax.
// ---------------------------------------------------------------------------

#define MAT (512 * 512)
#define NB 10

typedef __attribute__((ext_vector_type(16))) __bf16 v16bf;
typedef __attribute__((ext_vector_type(8)))  float  v8f;

union Frag16 { v16bf v; uint4 q[2]; };
union Pack8  { __bf16 h[8]; uint4 q; };

__device__ __forceinline__ float waveMax(float v) {
#pragma unroll
  for (int off = 16; off > 0; off >>= 1) v = fmaxf(v, __shfl_xor(v, off, 32));
  return v;
}
__device__ __forceinline__ float waveSum(float v) {
#pragma unroll
  for (int off = 16; off > 0; off >>= 1) v += __shfl_xor(v, off, 32);
  return v;
}

// ---------------------------------------------------------------------------
// prep1: bins[t] and per-row log-softmax of input_distros (512 x 10)
// ---------------------------------------------------------------------------
__global__ void prep1_kernel(const float* __restrict__ data,
                             const float* __restrict__ distros,
                             int* __restrict__ bins,
                             float* __restrict__ logleafs) {
  const int w = threadIdx.x;  // 0..511
  float d = data[w];
  int b = (int)floorf(d / 0.1f);
  bins[w] = b > (NB - 1) ? (NB - 1) : b;

  const float* row = distros + w * NB;
  float mx = row[0];
#pragma unroll
  for (int k = 1; k < NB; ++k) mx = fmaxf(mx, row[k]);
  float s = 0.f;
#pragma unroll
  for (int k = 0; k < NB; ++k) s += expf(row[k] - mx);
  float lse = mx + logf(s);
#pragma unroll
  for (int k = 0; k < NB; ++k) logleafs[w * NB + k] = row[k] - lse;
}

// ---------------------------------------------------------------------------
// prep2: lT[t][w] = logleafs[w][bins[t]]  and  colmax[t] = max_w lT[t][w]
// grid = 512 (t), block = 512 (w)
// ---------------------------------------------------------------------------
__global__ void prep2_kernel(const float* __restrict__ logleafs,
                             const int* __restrict__ bins,
                             float* __restrict__ lT,
                             float* __restrict__ colmax) {
  __shared__ float red[512];
  const int t = blockIdx.x, w = threadIdx.x;
  float v = logleafs[w * NB + bins[t]];
  lT[(size_t)t * 512 + w] = v;
  red[w] = v;
  __syncthreads();
  for (int s = 256; s > 0; s >>= 1) {
    if (w < s) red[w] = fmaxf(red[w], red[w + s]);
    __syncthreads();
  }
  if (w == 0) colmax[t] = red[0];
}

__global__ void zero_f32(float* __restrict__ p, int n) {
  int i = blockIdx.x * blockDim.x + threadIdx.x;
  if (i < n) p[i] = 0.0f;
}

// ---------------------------------------------------------------------------
// convert: M_t[i,j] = exp(lt[i] - s_t + W_t[i,j] - rowlse_i)   (bf16, <= 1)
// Written row-major (odd slot) or column-major (even slot) via an LDS-staged
// 32x512 strip so all global stores are 16B vectors.
// grid = (16 strips, 510 matrices), block = 256 (8 waves x 4 rows each)
// ---------------------------------------------------------------------------
__global__ void convert_kernel(const float* __restrict__ W,
                               const float* __restrict__ lT,
                               const float* __restrict__ colmax,
                               __bf16* __restrict__ outM,
                               float* __restrict__ scaleOut,
                               float* __restrict__ boundOut) {
  __shared__ __bf16 tile[32][512];  // 32 rows x full width (32 KB)
  const int m = blockIdx.y;  // matrix 0..509
  const int t = m + 1;       // time   1..510
  const int wave = threadIdx.x >> 5;
  const int lane = threadIdx.x & 31;
  const int i0 = blockIdx.x * 32;
  const float st = colmax[t];

  float lmax = 0.f;
#pragma unroll
  for (int rr = 0; rr < 4; ++rr) {
    const int ir = wave * 4 + rr;  // row within strip
    const int i = i0 + ir;
    const float* wr = W + ((size_t)t * 512 + i) * 512;
    float x[16];
    float mx = -INFINITY;
#pragma unroll
    for (int ii = 0; ii < 16; ++ii) {
      x[ii] = wr[lane + 32 * ii];
      mx = fmaxf(mx, x[ii]);
    }
    mx = waveMax(mx);
    float s = 0.f;
#pragma unroll
    for (int ii = 0; ii < 16; ++ii) s += expf(x[ii] - mx);
    s = waveSum(s);
    const float base = lT[(size_t)t * 512 + i] - st - (mx + logf(s));
#pragma unroll
    for (int ii = 0; ii < 16; ++ii) {
      float v = expf(base + x[ii]);
      lmax = fmaxf(lmax, v);
      tile[ir][lane + 32 * ii] = (__bf16)v;
    }
  }
  lmax = waveMax(lmax);
  if (lane == 0) atomicMax((unsigned int*)&boundOut[m], __float_as_uint(lmax));
  __syncthreads();

  if (m & 1) {  // odd slot -> left operand -> row-major
    __bf16* o = outM + (size_t)m * MAT + (size_t)i0 * 512;
    const __bf16* tf = &tile[0][0];
#pragma unroll
    for (int c = 0; c < 8; ++c) {
      int idx = (threadIdx.x + 256 * c) * 8;  // covers 32*512 elements
      *(uint4*)&o[idx] = *(const uint4*)&tf[idx];
    }
  } else {  // even slot -> right operand -> column-major: o[j*512 + i]
    __bf16* o = outM + (size_t)m * MAT;
#pragma unroll
    for (int c = 0; c < 2; ++c) {
      int j = threadIdx.x + 256 * c;  // column 0..511
#pragma unroll
      for (int g = 0; g < 4; ++g) {
        Pack8 pk;
#pragma unroll
        for (int e = 0; e < 8; ++e) pk.h[e] = tile[g * 8 + e][j];
        *(uint4*)&o[(size_t)j * 512 + i0 + g * 8] = pk.q;
      }
    }
  }
  if (blockIdx.x == 0 && threadIdx.x == 0) scaleOut[m] = st;
}

// ---------------------------------------------------------------------------
// pair_matmul: P_p = M[2p+1] x M[2p]  (512x512x512, bf16 in, f32 acc)
// L is row-major, R is column-major -> A and B staging paths are identical
// (global_load_b128 -> ds_store_b128 -> ds_load_b128, no scalar transposes).
// grid = (4, 4, pairs), block = 256 (8 wave32s), C tile 128x128, K-step 32.
// Each wave: 2x4 grid of 16x16 WMMA tiles -> 8 v_wmma per K-step.
// Output written row-major (p odd) or column-major (p even) for next level.
// ---------------------------------------------------------------------------
__global__ void __launch_bounds__(256) pair_matmul(
    const __bf16* __restrict__ inM, const float* __restrict__ inScale,
    const float* __restrict__ inBound, __bf16* __restrict__ outM,
    float* __restrict__ outScale, float* __restrict__ outBound) {
  __shared__ __bf16 As[128][40];  // padded: row stride 80B, 16B aligned
  __shared__ __bf16 Bs[128][40];  // Bs[n][k] (R is stored column-major)
  __shared__ float smax[8];

  const int tid = threadIdx.x;
  const int wave = tid >> 5, lane = tid & 31;
  const int waveM = wave & 3, waveN = wave >> 2;  // 4 x 2 wave grid
  const int mL = lane & 15, kh = lane >> 4;
  const int p = blockIdx.z;
  const int cRow0 = blockIdx.y * 128, cCol0 = blockIdx.x * 128;

  const __bf16* Lm = inM + (size_t)(2 * p + 1) * MAT;  // row-major
  const __bf16* Rm = inM + (size_t)(2 * p) * MAT;      // column-major
  const float D = fmaxf(inBound[2 * p + 1] * inBound[2 * p] * 512.0f, 1e-30f);

  v8f acc[2][4];
#pragma unroll
  for (int a = 0; a < 2; ++a)
#pragma unroll
    for (int b = 0; b < 4; ++b) acc[a][b] = (v8f)(0.0f);

  for (int kt = 0; kt < 16; ++kt) {
    const int k0 = kt * 32;
    // ---- global -> LDS, all 16B vectors, symmetric A/B paths ----
#pragma unroll
    for (int c = 0; c < 2; ++c) {
      int chunk = tid + 256 * c;  // 0..511
      int r = chunk >> 2;
      int kc = (chunk & 3) * 8;
      *(uint4*)&As[r][kc] =
          *(const uint4*)&Lm[(size_t)(cRow0 + r) * 512 + k0 + kc];
      *(uint4*)&Bs[r][kc] =
          *(const uint4*)&Rm[(size_t)(cCol0 + r) * 512 + k0 + kc];
    }
    if (kt + 1 < 16) {  // -> global_prefetch_b8 for the next K tile
      __builtin_prefetch(
          &Lm[(size_t)(cRow0 + (tid >> 2)) * 512 + k0 + 32 + (tid & 3) * 8], 0, 1);
      __builtin_prefetch(
          &Rm[(size_t)(cCol0 + (tid >> 2)) * 512 + k0 + 32 + (tid & 3) * 8], 0, 1);
    }
    __syncthreads();

    // ---- fragments from LDS (ds_load_b128) + WMMA ----
    Frag16 af[2];
#pragma unroll
    for (int mt = 0; mt < 2; ++mt) {
      int row = waveM * 32 + mt * 16 + mL;
      af[mt].q[0] = *(const uint4*)&As[row][kh * 8];       // K 0..7 / 8..15
      af[mt].q[1] = *(const uint4*)&As[row][16 + kh * 8];  // K 16..23 / 24..31
    }
#pragma unroll
    for (int nt = 0; nt < 4; ++nt) {
      Frag16 bfv;
      int nrow = waveN * 64 + nt * 16 + mL;
      bfv.q[0] = *(const uint4*)&Bs[nrow][kh * 8];
      bfv.q[1] = *(const uint4*)&Bs[nrow][16 + kh * 8];
#pragma unroll
      for (int mt = 0; mt < 2; ++mt)
        acc[mt][nt] = __builtin_amdgcn_wmma_f32_16x16x32_bf16(
            false, af[mt].v, false, bfv.v, (short)0, acc[mt][nt], false, false);
    }
    __syncthreads();
  }

  // ---- epilogue: normalize by bound product, write in consumer layout ----
  const float invD = 1.0f / D;
  float lmax = 0.f;
  __bf16* O = outM + (size_t)p * MAT;
  if (p & 1) {  // next level uses this as left operand -> row-major
#pragma unroll
    for (int mt = 0; mt < 2; ++mt)
#pragma unroll
      for (int nt = 0; nt < 4; ++nt)
#pragma unroll
        for (int r = 0; r < 8; ++r) {
          float v = acc[mt][nt][r] * invD;
          lmax = fmaxf(lmax, v);
          int row = cRow0 + waveM * 32 + mt * 16 + (lane >> 4) * 8 + r;
          int col = cCol0 + waveN * 64 + nt * 16 + (lane & 15);
          O[(size_t)row * 512 + col] = (__bf16)v;
        }
  } else {  // right operand -> column-major; 8 rows pack into one 16B store
#pragma unroll
    for (int mt = 0; mt < 2; ++mt)
#pragma unroll
      for (int nt = 0; nt < 4; ++nt) {
        Pack8 pk;
#pragma unroll
        for (int r = 0; r < 8; ++r) {
          float v = acc[mt][nt][r] * invD;
          lmax = fmaxf(lmax, v);
          pk.h[r] = (__bf16)v;
        }
        int row0 = cRow0 + waveM * 32 + mt * 16 + (lane >> 4) * 8;
        int col = cCol0 + waveN * 64 + nt * 16 + (lane & 15);
        *(uint4*)&O[(size_t)col * 512 + row0] = pk.q;
      }
  }
  lmax = waveMax(lmax);
  if (lane == 0) smax[wave] = lmax;
  __syncthreads();
  if (tid == 0) {
    float mv = smax[0];
#pragma unroll
    for (int i2 = 1; i2 < 8; ++i2) mv = fmaxf(mv, smax[i2]);
    atomicMax((unsigned int*)&outBound[p], __float_as_uint(mv));
    if (blockIdx.x == 0 && blockIdx.y == 0)
      outScale[p] = inScale[2 * p] + inScale[2 * p + 1] + logf(D);
  }
}

// ---------------------------------------------------------------------------
// transpose_bf16: dst[i*512+j] = src[j*512+i]   (carry slot parity flip)
// grid (16,16), block (32,8)
// ---------------------------------------------------------------------------
__global__ void transpose_bf16(const __bf16* __restrict__ src,
                               __bf16* __restrict__ dst) {
  __shared__ __bf16 tl[32][33];
  const int bx = blockIdx.x * 32, by = blockIdx.y * 32;
#pragma unroll
  for (int yy = 0; yy < 32; yy += 8)
    tl[threadIdx.y + yy][threadIdx.x] =
        src[(size_t)(by + threadIdx.y + yy) * 512 + bx + threadIdx.x];
  __syncthreads();
#pragma unroll
  for (int yy = 0; yy < 32; yy += 8)
    dst[(size_t)(bx + threadIdx.y + yy) * 512 + by + threadIdx.x] =
        tl[threadIdx.x][threadIdx.y + yy];
}

// ---------------------------------------------------------------------------
// finalize: y = G_hat * a0_hat; out = exp(log y + S + s0 + lse(l[:,-1]))
// G_hat is stored column-major (slot 0 is even) -> coalesced GEMV reads.
// one block, 512 threads
// ---------------------------------------------------------------------------
__global__ void finalize_kernel(const __bf16* __restrict__ Gcm,
                                const float* __restrict__ Sf,
                                const float* __restrict__ lT,
                                const float* __restrict__ colmax,
                                float* __restrict__ out) {
  __shared__ float a0[512];
  __shared__ float red[512];
  const int w = threadIdx.x;
  const float s0 = colmax[0];
  a0[w] = expf(lT[w] - s0);

  float v = lT[(size_t)511 * 512 + w];
  red[w] = v;
  __syncthreads();
  for (int s = 256; s > 0; s >>= 1) {
    if (w < s) red[w] = fmaxf(red[w], red[w + s]);
    __syncthreads();
  }
  float mx = red[0];
  __syncthreads();
  red[w] = expf(v - mx);
  __syncthreads();
  for (int s = 256; s > 0; s >>= 1) {
    if (w < s) red[w] += red[w + s];
    __syncthreads();
  }
  float lseLast = mx + logf(red[0]);
  __syncthreads();

  float sum = 0.f;
  for (int j = 0; j < 512; ++j)  // Gcm[j*512 + w] = G[w][j], coalesced in w
    sum += (float)Gcm[(size_t)j * 512 + w] * a0[j];
  out[w] = expf(logf(fmaxf(sum, 1e-37f)) + Sf[0] + s0 + lseLast);
}

// ---------------------------------------------------------------------------
// host driver
// ---------------------------------------------------------------------------
extern "C" void kernel_launch(void* const* d_in, const int* in_sizes, int n_in,
                              void* d_out, int out_size, void* d_ws,
                              size_t ws_size, hipStream_t stream) {
  (void)in_sizes; (void)n_in; (void)out_size; (void)ws_size;
  const float* data    = (const float*)d_in[0];  // (1,512)
  const float* distros = (const float*)d_in[1];  // (512,10)
  const float* W       = (const float*)d_in[2];  // (511,512,512)
  float* out = (float*)d_out;

  // ---- workspace layout (~402 MB) ----
  char* p = (char*)d_ws;
  __bf16* matsA = (__bf16*)p; p += (size_t)510 * MAT * 2;  // levels 0,2,4,6,8
  __bf16* matsB = (__bf16*)p; p += (size_t)255 * MAT * 2;  // levels 1,3,5,7,9
  float* lT       = (float*)p; p += (size_t)MAT * 4;       // l^T (t-major)
  float* colmax   = (float*)p; p += 512 * 4;
  float* logleafs = (float*)p; p += 512 * NB * 4;
  int*   bins     = (int*)p;   p += 512 * 4;
  float* scaleA   = (float*)p; p += 512 * 4;
  float* boundA   = (float*)p; p += 512 * 4;
  float* scaleB   = (float*)p; p += 512 * 4;
  float* boundB   = (float*)p; p += 512 * 4;

  prep1_kernel<<<1, 512, 0, stream>>>(data, distros, bins, logleafs);
  prep2_kernel<<<512, 512, 0, stream>>>(logleafs, bins, lT, colmax);
  zero_f32<<<2, 256, 0, stream>>>(boundA, 510);
  convert_kernel<<<dim3(16, 510), 256, 0, stream>>>(W, lT, colmax, matsA,
                                                    scaleA, boundA);

  __bf16* mats[2] = {matsA, matsB};
  float* scl[2] = {scaleA, scaleB};
  float* bnd[2] = {boundA, boundB};
  int n = 510, cur = 0;
  while (n > 1) {  // 510->255->128->64->32->16->8->4->2->1  (9 levels)
    int nxt = cur ^ 1;
    int pairs = n >> 1;
    int carry = n & 1;
    int nOut = pairs + carry;
    zero_f32<<<2, 256, 0, stream>>>(bnd[nxt], nOut);
    pair_matmul<<<dim3(4, 4, pairs), 256, 0, stream>>>(
        mats[cur], scl[cur], bnd[cur], mats[nxt], scl[nxt], bnd[nxt]);
    if (carry) {  // pass the odd matrix through (layout parity may flip)
      const __bf16* src = mats[cur] + (size_t)(n - 1) * MAT;
      __bf16* dst = mats[nxt] + (size_t)(nOut - 1) * MAT;
      if (((n - 1) & 1) != ((nOut - 1) & 1))
        transpose_bf16<<<dim3(16, 16), dim3(32, 8), 0, stream>>>(src, dst);
      else
        hipMemcpyAsync(dst, src, (size_t)MAT * 2, hipMemcpyDeviceToDevice,
                       stream);
      hipMemcpyAsync(scl[nxt] + (nOut - 1), scl[cur] + (n - 1), 4,
                     hipMemcpyDeviceToDevice, stream);
      hipMemcpyAsync(bnd[nxt] + (nOut - 1), bnd[cur] + (n - 1), 4,
                     hipMemcpyDeviceToDevice, stream);
    }
    n = nOut;
    cur = nxt;
  }
  finalize_kernel<<<1, 512, 0, stream>>>(mats[cur], scl[cur], lT, colmax, out);
}
